// RelationAwareMultiheadAttention_50062138802732
// MI455X (gfx1250) — matrix operand order repaired and checked by reading, compiled, not run on hardware
//
#include <hip/hip_runtime.h>

// ---------------------------------------------------------------------------
// Relation-aware MHA for gfx1250 (MI455X). All GEMM-shaped work uses
// v_wmma_f32_16x16x32_bf16 (wave32, 16x16 tiles, f32 accumulate).
//
// Memory strategy (23.3 TB/s HBM, 192 MB global L2):
//  - relations_k/relations_v (1 GiB total) are read exactly once -> NT loads
//    so they stream through without evicting the score matrix.
//  - S/P score matrix (128 MiB) is touched by 5 consecutive kernels and fits
//    in the 192 MB L2 -> default (RT) policy keeps it resident.
//  - final outputs are written once -> NT stores.
//
// Workspace layout (needs ~148 MiB):
//   [0,4M)      q_h  bf16 (B,H,S,D)   (pre-scaled by D^-0.5)
//   [4M,8M)     k_h  bf16
//   [8M,12M)    v_h  bf16
//   [12M,140M)  S/P f32 (B,H,Sq,Sa)   scores, softmaxed in place
//   [140M,148M) attn_pre f32 (B,Sq,E) pre-output-projection
// ---------------------------------------------------------------------------

#define BATCH 2
#define SEQ   1024
#define EMB   1024
#define NHEAD 16
#define HDIM  64

typedef __attribute__((ext_vector_type(16))) __bf16 v16bf;
typedef __attribute__((ext_vector_type(8)))  float  v8f;
typedef __attribute__((ext_vector_type(4)))  float  f4v;

union BFrag {
  v16bf v;
  uint4 q[2];
  unsigned short u[16];
};

__device__ __forceinline__ unsigned short f2bf(float f) {
  unsigned int x = __float_as_uint(f);
  x += 0x7FFFu + ((x >> 16) & 1u);          // round-to-nearest-even
  return (unsigned short)(x >> 16);
}

#if __has_builtin(__builtin_amdgcn_cvt_pk_bf16_f32)
__device__ __forceinline__ unsigned int pk2(float a, float b) {
  auto r = __builtin_amdgcn_cvt_pk_bf16_f32(a, b);   // v_cvt_pk_bf16_f32
  unsigned int u = 0;
  __builtin_memcpy(&u, &r, sizeof(r) < 4 ? sizeof(r) : 4);
  return u;
}
#else
__device__ __forceinline__ unsigned int pk2(float a, float b) {
  return (unsigned int)f2bf(a) | ((unsigned int)f2bf(b) << 16);
}
#endif

__device__ __forceinline__ float bf2f(unsigned short u) {
  return __uint_as_float(((unsigned int)u) << 16);
}
__device__ __forceinline__ v8f wmma_bf16(const BFrag& a, const BFrag& b, v8f c) {
  return __builtin_amdgcn_wmma_f32_16x16x32_bf16(
      false, a.v, false, b.v, (short)0, c, false, false);
}

// ---------------------------------------------------------------------------
// Generic  C = A(MxK) @ W(NxK)^T + bias.  mode 0: f32 row-major out (NT).
// mode 1: bf16 head-major out[((b*H+h)*S+s)*D+d], value scaled by `scale`.
// Block tile 128x128x64, 8 waves, each wave 64x32 (4x2 WMMA tiles).
// ---------------------------------------------------------------------------
__global__ __launch_bounds__(256) void ramha_gemm_xw(
    const float* __restrict__ A, const float* __restrict__ W,
    const float* __restrict__ bias, float* __restrict__ outF,
    unsigned short* __restrict__ outBF, int M, int N, int K,
    float scale, int mode)
{
  __shared__ __align__(16) unsigned short sA[128 * 72];
  __shared__ __align__(16) unsigned short sB[128 * 72];
  const int m0 = blockIdx.y * 128, n0 = blockIdx.x * 128;
  const int tid = threadIdx.x;
  const int lane = tid & 31, wv = tid >> 5;
  const int wr = wv >> 2, wc = wv & 3;          // 2 x 4 wave grid
  const int lx = lane & 15, hh = lane >> 4;

  v8f acc[4][2];
  for (int i = 0; i < 4; ++i)
    for (int j = 0; j < 2; ++j)
      for (int r = 0; r < 8; ++r) acc[i][j][r] = 0.0f;

  const int lrow = tid >> 1;
  const int lcol = (tid & 1) * 32;

  for (int k0 = 0; k0 < K; k0 += 64) {
    __syncthreads();
    {  // stage A tile, f32 -> bf16
      const float* src = A + (size_t)(m0 + lrow) * K + k0 + lcol;
      unsigned short* dst = &sA[lrow * 72 + lcol];
      for (int j = 0; j < 4; ++j) {
        float4 f0 = *(const float4*)(src + 8 * j);
        float4 f1 = *(const float4*)(src + 8 * j + 4);
        uint4 u;
        u.x = pk2(f0.x, f0.y); u.y = pk2(f0.z, f0.w);
        u.z = pk2(f1.x, f1.y); u.w = pk2(f1.z, f1.w);
        *(uint4*)(dst + 8 * j) = u;
      }
    }
    {  // stage W tile (row n = output col, k contiguous)
      const float* src = W + (size_t)(n0 + lrow) * K + k0 + lcol;
      unsigned short* dst = &sB[lrow * 72 + lcol];
      for (int j = 0; j < 4; ++j) {
        float4 f0 = *(const float4*)(src + 8 * j);
        float4 f1 = *(const float4*)(src + 8 * j + 4);
        uint4 u;
        u.x = pk2(f0.x, f0.y); u.y = pk2(f0.z, f0.w);
        u.z = pk2(f1.x, f1.y); u.w = pk2(f1.z, f1.w);
        *(uint4*)(dst + 8 * j) = u;
      }
    }
    __syncthreads();
    for (int ks = 0; ks < 64; ks += 32) {
      BFrag a[4], bf[2];
      for (int mt = 0; mt < 4; ++mt) {           // A frag: lane = M row
        const int off = (wr * 64 + mt * 16 + lx) * 72 + ks + hh * 16;
        a[mt].q[0] = *(const uint4*)&sA[off];
        a[mt].q[1] = *(const uint4*)&sA[off + 8];
      }
      for (int nt = 0; nt < 2; ++nt) {           // B frag: lane = N col
        const int off = (wc * 32 + nt * 16 + lx) * 72 + ks + hh * 16;
        bf[nt].q[0] = *(const uint4*)&sB[off];
        bf[nt].q[1] = *(const uint4*)&sB[off + 8];
      }
      for (int mt = 0; mt < 4; ++mt)
        for (int nt = 0; nt < 2; ++nt)
          acc[mt][nt] = wmma_bf16(a[mt], bf[nt], acc[mt][nt]);
    }
  }
  for (int mt = 0; mt < 4; ++mt)
    for (int nt = 0; nt < 2; ++nt) {
      const int colg = n0 + wc * 32 + nt * 16 + lx;
      const float bv = bias[colg];
      const int rowb = m0 + wr * 64 + mt * 16 + hh * 8;
      for (int r = 0; r < 8; ++r) {
        const float val = acc[mt][nt][r] + bv;
        const int rowg = rowb + r;
        if (mode == 0) {
          __builtin_nontemporal_store(val, &outF[(size_t)rowg * N + colg]);
        } else {
          const int bb = rowg >> 10, ss = rowg & 1023;
          const int hd = colg >> 6,  dd = colg & 63;
          outBF[((size_t)(bb * NHEAD + hd) * SEQ + ss) * HDIM + dd] =
              f2bf(val * scale);
        }
      }
    }
}

// ---------------------------------------------------------------------------
// Per (b, qpos): S[b,h,qpos,:] = Q16(16x64) @ relk[b,qpos]^T + mask.
// M = heads(16), K = D(64), N = key positions. rel_k streamed once (NT).
// ---------------------------------------------------------------------------
__global__ __launch_bounds__(128) void ramha_rel_score(
    const unsigned short* __restrict__ qh, const float* __restrict__ relk,
    const float* __restrict__ amask, const unsigned char* __restrict__ kpad,
    float* __restrict__ S)
{
  const int qpos = blockIdx.x, b = blockIdx.y;
  __shared__ __align__(16) unsigned short sQ[16 * 72];
  const int tid = threadIdx.x;
  {
    const int row = tid >> 3, chunk = (tid & 7) * 8;
    const unsigned short* src =
        qh + ((size_t)(b * NHEAD + row) * SEQ + qpos) * HDIM + chunk;
    *(uint4*)&sQ[row * 72 + chunk] = *(const uint4*)src;
  }
  __syncthreads();
  const int lane = tid & 31, wv = tid >> 5;
  const int lx = lane & 15, hh = lane >> 4;
  BFrag a0, a1;
  {
    const int off = lx * 72 + hh * 16;
    a0.q[0] = *(const uint4*)&sQ[off];
    a0.q[1] = *(const uint4*)&sQ[off + 8];
    a1.q[0] = *(const uint4*)&sQ[off + 32];
    a1.q[1] = *(const uint4*)&sQ[off + 40];
  }
  const float* rk = relk + ((size_t)b * SEQ + qpos) * SEQ * HDIM;
  const float* am = amask + ((size_t)b * SEQ + qpos) * SEQ;
  const unsigned char* kp = kpad + (size_t)b * SEQ;
  float* Sbase = S + (((size_t)b * NHEAD) * SEQ + qpos) * SEQ;

  for (int nt = wv; nt < SEQ / 16; nt += 4) {
    const int n = nt * 16 + lx;                  // key position (N)
    const float* rrow = rk + (size_t)n * HDIM;   // 64 contiguous d values
    BFrag b0, b1;
    for (int j = 0; j < 4; ++j) {
      f4v f = __builtin_nontemporal_load((const f4v*)(rrow + hh * 16 + 4 * j));
      b0.u[4 * j + 0] = f2bf(f.x); b0.u[4 * j + 1] = f2bf(f.y);
      b0.u[4 * j + 2] = f2bf(f.z); b0.u[4 * j + 3] = f2bf(f.w);
      f4v g =
          __builtin_nontemporal_load((const f4v*)(rrow + 32 + hh * 16 + 4 * j));
      b1.u[4 * j + 0] = f2bf(g.x); b1.u[4 * j + 1] = f2bf(g.y);
      b1.u[4 * j + 2] = f2bf(g.z); b1.u[4 * j + 3] = f2bf(g.w);
    }
    v8f c;
    for (int r = 0; r < 8; ++r) c[r] = 0.0f;
    c = wmma_bf16(a0, b0, c);
    c = wmma_bf16(a1, b1, c);
    const float mv = am[n] + (kp[n] ? -__builtin_inff() : 0.0f);
    for (int r = 0; r < 8; ++r)
      Sbase[(size_t)(hh * 8 + r) * (SEQ * SEQ) + n] = c[r] + mv;
  }
}

// ---------------------------------------------------------------------------
// Per (b,h): S += Q @ K^T. 128x128 tile, K-dim = 64. Fragments loaded
// straight from global (128 KB slices live in the 192 MB L2).
// ---------------------------------------------------------------------------
__global__ __launch_bounds__(256) void ramha_qk_acc(
    const unsigned short* __restrict__ qh, const unsigned short* __restrict__ kh,
    float* __restrict__ S)
{
  const int nt0 = blockIdx.x, qt = blockIdx.y, bh = blockIdx.z;
  const unsigned short* Q = qh + (size_t)bh * SEQ * HDIM;
  const unsigned short* Kk = kh + (size_t)bh * SEQ * HDIM;
  float* Sm = S + (size_t)bh * SEQ * SEQ;
  const int tid = threadIdx.x;
  const int lane = tid & 31, wv = tid >> 5;
  const int wr = wv >> 2, wc = wv & 3;
  const int lx = lane & 15, hh = lane >> 4;
  const int r0 = qt * 128 + wr * 64;
  const int c0 = nt0 * 128 + wc * 32;

  v8f acc[4][2];
  for (int mt = 0; mt < 4; ++mt)
    for (int nt = 0; nt < 2; ++nt) {
      const int rowb = r0 + mt * 16 + hh * 8;
      const int col = c0 + nt * 16 + lx;
      for (int r = 0; r < 8; ++r)
        acc[mt][nt][r] = Sm[(size_t)(rowb + r) * SEQ + col];
    }
  for (int ks = 0; ks < 64; ks += 32) {
    BFrag a[4], bf[2];
    for (int mt = 0; mt < 4; ++mt) {
      const int off = (r0 + mt * 16 + lx) * HDIM + ks + hh * 16;
      a[mt].q[0] = *(const uint4*)(Q + off);
      a[mt].q[1] = *(const uint4*)(Q + off + 8);
    }
    for (int nt = 0; nt < 2; ++nt) {
      const int off = (c0 + nt * 16 + lx) * HDIM + ks + hh * 16;
      bf[nt].q[0] = *(const uint4*)(Kk + off);
      bf[nt].q[1] = *(const uint4*)(Kk + off + 8);
    }
    for (int mt = 0; mt < 4; ++mt)
      for (int nt = 0; nt < 2; ++nt)
        acc[mt][nt] = wmma_bf16(a[mt], bf[nt], acc[mt][nt]);
  }
  for (int mt = 0; mt < 4; ++mt)
    for (int nt = 0; nt < 2; ++nt) {
      const int rowb = r0 + mt * 16 + hh * 8;
      const int col = c0 + nt * 16 + lx;
      for (int r = 0; r < 8; ++r)
        Sm[(size_t)(rowb + r) * SEQ + col] = acc[mt][nt][r];
    }
}

// ---------------------------------------------------------------------------
// Row softmax in place. One 128-thread block per row, 8 cols/thread.
// ---------------------------------------------------------------------------
__global__ __launch_bounds__(128) void ramha_softmax(float* __restrict__ S)
{
  float* row = S + (size_t)blockIdx.x * SEQ;
  const int t = threadIdx.x, lane = t & 31;
  float4 v0 = *(const float4*)(row + t * 8);
  float4 v1 = *(const float4*)(row + t * 8 + 4);
  float m = fmaxf(fmaxf(fmaxf(v0.x, v0.y), fmaxf(v0.z, v0.w)),
                  fmaxf(fmaxf(v1.x, v1.y), fmaxf(v1.z, v1.w)));
  for (int off = 16; off > 0; off >>= 1) m = fmaxf(m, __shfl_xor(m, off, 32));
  __shared__ float red[8];
  if (lane == 0) red[t >> 5] = m;
  __syncthreads();
  m = fmaxf(fmaxf(red[0], red[1]), fmaxf(red[2], red[3]));
  v0.x = __expf(v0.x - m); v0.y = __expf(v0.y - m);
  v0.z = __expf(v0.z - m); v0.w = __expf(v0.w - m);
  v1.x = __expf(v1.x - m); v1.y = __expf(v1.y - m);
  v1.z = __expf(v1.z - m); v1.w = __expf(v1.w - m);
  float s = v0.x + v0.y + v0.z + v0.w + v1.x + v1.y + v1.z + v1.w;
  for (int off = 16; off > 0; off >>= 1) s += __shfl_xor(s, off, 32);
  if (lane == 0) red[4 + (t >> 5)] = s;
  __syncthreads();
  s = red[4] + red[5] + red[6] + red[7];
  const float inv = 1.0f / s;
  v0.x *= inv; v0.y *= inv; v0.z *= inv; v0.w *= inv;
  v1.x *= inv; v1.y *= inv; v1.z *= inv; v1.w *= inv;
  *(float4*)(row + t * 8) = v0;
  *(float4*)(row + t * 8 + 4) = v1;
}

// ---------------------------------------------------------------------------
// Per (b,h, 128-row q-tile): attn_pre[b,q,h*64+d] = P @ V.
// P f32 -> bf16 staged in LDS; V transposed into LDS for B-fragment layout.
// ---------------------------------------------------------------------------
__global__ __launch_bounds__(256) void ramha_pv(
    const float* __restrict__ P, const unsigned short* __restrict__ vh,
    float* __restrict__ attn_pre)
{
  const int qt = blockIdx.x, bh = blockIdx.y;
  const int b = bh >> 4, h = bh & 15;
  const float* Pm = P + (size_t)bh * SEQ * SEQ + (size_t)qt * 128 * SEQ;
  const unsigned short* V = vh + (size_t)bh * SEQ * HDIM;
  __shared__ __align__(16) unsigned short sP[128 * 72];
  __shared__ __align__(16) unsigned short sVT[64 * 72];
  const int tid = threadIdx.x;
  const int lane = tid & 31, wv = tid >> 5;
  const int wr = wv >> 1, wc = wv & 1;          // 4 x 2 wave grid, 32x32/wave
  const int lx = lane & 15, hh = lane >> 4;

  v8f acc[2][2];
  for (int i = 0; i < 2; ++i)
    for (int j = 0; j < 2; ++j)
      for (int r = 0; r < 8; ++r) acc[i][j][r] = 0.0f;

  const int prow = tid >> 1, pchunk = (tid & 1) * 32;
  const int vrow = tid >> 2, vchunk = (tid & 3) * 16;

  for (int k0 = 0; k0 < SEQ; k0 += 64) {
    __syncthreads();
    {  // stage P (f32 -> bf16)
      const float* src = Pm + (size_t)prow * SEQ + k0 + pchunk;
      unsigned short* dst = &sP[prow * 72 + pchunk];
      for (int j = 0; j < 4; ++j) {
        float4 f0 = *(const float4*)(src + 8 * j);
        float4 f1 = *(const float4*)(src + 8 * j + 4);
        uint4 u;
        u.x = pk2(f0.x, f0.y); u.y = pk2(f0.z, f0.w);
        u.z = pk2(f1.x, f1.y); u.w = pk2(f1.z, f1.w);
        *(uint4*)(dst + 8 * j) = u;
      }
    }
    {  // stage V transposed: sVT[d][kpos]
      const unsigned short* src = V + (size_t)(k0 + vrow) * HDIM + vchunk;
      unsigned short tmp[16];
      *(uint4*)tmp = *(const uint4*)src;
      *(uint4*)(tmp + 8) = *(const uint4*)(src + 8);
      for (int j = 0; j < 16; ++j) sVT[(vchunk + j) * 72 + vrow] = tmp[j];
    }
    __syncthreads();
    for (int ks = 0; ks < 64; ks += 32) {
      BFrag a[2], bf[2];
      for (int mt = 0; mt < 2; ++mt) {
        const int off = (wr * 32 + mt * 16 + lx) * 72 + ks + hh * 16;
        a[mt].q[0] = *(const uint4*)&sP[off];
        a[mt].q[1] = *(const uint4*)&sP[off + 8];
      }
      for (int nt = 0; nt < 2; ++nt) {
        const int off = (wc * 32 + nt * 16 + lx) * 72 + ks + hh * 16;
        bf[nt].q[0] = *(const uint4*)&sVT[off];
        bf[nt].q[1] = *(const uint4*)&sVT[off + 8];
      }
      for (int mt = 0; mt < 2; ++mt)
        for (int nt = 0; nt < 2; ++nt)
          acc[mt][nt] = wmma_bf16(a[mt], bf[nt], acc[mt][nt]);
    }
  }
  for (int mt = 0; mt < 2; ++mt)
    for (int nt = 0; nt < 2; ++nt) {
      const int q0 = qt * 128 + wr * 32 + mt * 16 + hh * 8;
      const int d = wc * 32 + nt * 16 + lx;
      for (int r = 0; r < 8; ++r)
        attn_pre[((size_t)b * SEQ + q0 + r) * EMB + h * HDIM + d] =
            acc[mt][nt][r];
    }
}

// ---------------------------------------------------------------------------
// Per (b, qpos): attn_pre[b,qpos,h*64+d] += P16(16x1024) @ relv[b,qpos]
// (M=heads, K=Sa, N=D), plus head-averaged attention weights output.
// rel_v streamed once (NT loads + prefetch ahead).
// ---------------------------------------------------------------------------
__global__ __launch_bounds__(128) void ramha_relv_avg(
    const float* __restrict__ P, const float* __restrict__ relv,
    float* __restrict__ attn_pre, float* __restrict__ avg_out)
{
  const int qpos = blockIdx.x, b = blockIdx.y;
  __shared__ __align__(16) unsigned short sP[16 * 1032];
  const int tid = threadIdx.x;
  {  // stage P16 (16 heads x 1024) f32 -> bf16
    const int row = tid >> 3, chunk = (tid & 7) * 128;
    const float* src =
        P + ((size_t)(b * NHEAD + row) * SEQ + qpos) * SEQ + chunk;
    unsigned short* dst = &sP[row * 1032 + chunk];
    for (int j = 0; j < 16; ++j) {
      float4 f0 = *(const float4*)(src + 8 * j);
      float4 f1 = *(const float4*)(src + 8 * j + 4);
      uint4 u;
      u.x = pk2(f0.x, f0.y); u.y = pk2(f0.z, f0.w);
      u.z = pk2(f1.x, f1.y); u.w = pk2(f1.z, f1.w);
      *(uint4*)(dst + 8 * j) = u;
    }
  }
  __syncthreads();
  const int lane = tid & 31, wv = tid >> 5;
  const int lx = lane & 15, hh = lane >> 4;
  const int n0 = wv * 16;                       // d-tile per wave
  const float* rv = relv + ((size_t)b * SEQ + qpos) * SEQ * HDIM;
  v8f acc;
  for (int r = 0; r < 8; ++r) acc[r] = 0.0f;

  for (int k0 = 0; k0 < SEQ; k0 += 32) {
    if (k0 + 32 < SEQ)
      __builtin_prefetch(rv + (size_t)(k0 + 32) * HDIM + n0, 0, 0);
    BFrag a, bf;
    const int off = lx * 1032 + k0 + hh * 16;
    a.q[0] = *(const uint4*)&sP[off];
    a.q[1] = *(const uint4*)&sP[off + 8];
    const int d = n0 + lx;
    for (int vv = 0; vv < 8; ++vv) {
      const int kp = k0 + hh * 16 + 2 * vv;
      bf.u[2 * vv + 0] =
          f2bf(__builtin_nontemporal_load(rv + (size_t)kp * HDIM + d));
      bf.u[2 * vv + 1] =
          f2bf(__builtin_nontemporal_load(rv + (size_t)(kp + 1) * HDIM + d));
    }
    acc = wmma_bf16(a, bf, acc);
  }
  float* ap = attn_pre + ((size_t)b * SEQ + qpos) * EMB;
  for (int r = 0; r < 8; ++r) {
    const int hrow = hh * 8 + r;
    ap[hrow * HDIM + n0 + lx] += acc[r];
  }
  // head-averaged attention weights (second output)
  float* av = avg_out + ((size_t)b * SEQ + qpos) * SEQ;
  const int c0 = tid * 8;
  float sums[8] = {0, 0, 0, 0, 0, 0, 0, 0};
  for (int row = 0; row < 16; ++row)
    for (int j = 0; j < 8; ++j) sums[j] += bf2f(sP[row * 1032 + c0 + j]);
  for (int j = 0; j < 8; ++j)
    __builtin_nontemporal_store(sums[j] * (1.0f / 16.0f), &av[c0 + j]);
}

// ---------------------------------------------------------------------------
extern "C" void kernel_launch(void* const* d_in, const int* in_sizes, int n_in,
                              void* d_out, int out_size, void* d_ws,
                              size_t ws_size, hipStream_t stream)
{
  (void)in_sizes; (void)n_in; (void)out_size; (void)ws_size;
  const float* query = (const float*)d_in[0];
  const float* key   = (const float*)d_in[1];
  const float* value = (const float*)d_in[2];
  const float* relk  = (const float*)d_in[3];
  const float* relv  = (const float*)d_in[4];
  const float* amask = (const float*)d_in[5];
  const unsigned char* kpad = (const unsigned char*)d_in[6];
  const float* q_w = (const float*)d_in[7];
  const float* q_b = (const float*)d_in[8];
  const float* k_w = (const float*)d_in[9];
  const float* k_b = (const float*)d_in[10];
  const float* v_w = (const float*)d_in[11];
  const float* v_b = (const float*)d_in[12];
  const float* o_w = (const float*)d_in[13];
  const float* o_b = (const float*)d_in[14];

  char* ws = (char*)d_ws;
  unsigned short* qh = (unsigned short*)(ws);
  unsigned short* kh = (unsigned short*)(ws + ((size_t)4 << 20));
  unsigned short* vh = (unsigned short*)(ws + ((size_t)8 << 20));
  float* S        = (float*)(ws + ((size_t)12 << 20));
  float* attn_pre = (float*)(ws + ((size_t)140 << 20));
  float* out_attn = (float*)d_out;
  float* out_avg  = out_attn + (size_t)BATCH * SEQ * EMB;

  const float scaling = 0.125f;  // HEAD_DIM^-0.5
  const dim3 gGemm(EMB / 128, (BATCH * SEQ) / 128);  // (8, 16)

  ramha_gemm_xw<<<gGemm, 256, 0, stream>>>(query, q_w, q_b, nullptr, qh,
                                           BATCH * SEQ, EMB, EMB, scaling, 1);
  ramha_gemm_xw<<<gGemm, 256, 0, stream>>>(key, k_w, k_b, nullptr, kh,
                                           BATCH * SEQ, EMB, EMB, 1.0f, 1);
  ramha_gemm_xw<<<gGemm, 256, 0, stream>>>(value, v_w, v_b, nullptr, vh,
                                           BATCH * SEQ, EMB, EMB, 1.0f, 1);
  ramha_rel_score<<<dim3(SEQ, BATCH), 128, 0, stream>>>(qh, relk, amask, kpad, S);
  ramha_qk_acc<<<dim3(SEQ / 128, SEQ / 128, BATCH * NHEAD), 256, 0, stream>>>(
      qh, kh, S);
  ramha_softmax<<<BATCH * NHEAD * SEQ, 128, 0, stream>>>(S);
  ramha_pv<<<dim3(SEQ / 128, BATCH * NHEAD), 256, 0, stream>>>(S, vh, attn_pre);
  ramha_relv_avg<<<dim3(SEQ, BATCH), 128, 0, stream>>>(S, relv, attn_pre,
                                                       out_avg);
  ramha_gemm_xw<<<gGemm, 256, 0, stream>>>(attn_pre, o_w, o_b, out_attn,
                                           nullptr, BATCH * SEQ, EMB, EMB,
                                           1.0f, 0);
}
